// Q2_54116587929805
// MI455X (gfx1250) — compile-verified, matrix operand order checked by reading, not run on hardware
//
#include <hip/hip_runtime.h>
#include <hip/hip_bf16.h>
#include <math.h>

typedef __attribute__((ext_vector_type(16))) _Float16 v16h;
typedef __attribute__((ext_vector_type(8)))  float    v8f;

#define B_SZ   1024
#define K_SZ   512

__device__ __forceinline__ float sigm(float x) { return 1.0f / (1.0f + __expf(-x)); }

// ---------------------------------------------------------------------------
// Kernel 1: action path.  One block (256 threads) per batch row.
// pre[129] -> MLP(129,256,256) -> LSTM(256) -> MLP(256,128,64) -> actemb
// ---------------------------------------------------------------------------
__global__ void __launch_bounds__(256) action_path_kernel(
    const float* __restrict__ states2, const float* __restrict__ h0,
    const float* __restrict__ c0, const int* __restrict__ a0,
    const int* __restrict__ a1, const float* __restrict__ node_emb,
    const float* __restrict__ W1a, const float* __restrict__ b1a,
    const float* __restrict__ W1b, const float* __restrict__ b1b,
    const float* __restrict__ W3a, const float* __restrict__ b3a,
    const float* __restrict__ W3b, const float* __restrict__ b3b,
    const float* __restrict__ W_ih, const float* __restrict__ W_hh,
    const float* __restrict__ b_ih, const float* __restrict__ b_hh,
    float* __restrict__ out_h1, float* __restrict__ out_c1,
    float* __restrict__ actemb)
{
    __shared__ float s_pre[132];
    __shared__ float s_x[256];
    __shared__ float s_emb[256];
    __shared__ float s_h0[256];
    __shared__ float s_h1[256];
    __shared__ float s_t[128];

    const int b = blockIdx.x;
    const int j = threadIdx.x;

    if (j < 64)        s_pre[j]   = node_emb[(long long)a0[b] * 64 + j];
    else if (j < 128)  s_pre[j]   = node_emb[(long long)a1[b] * 64 + (j - 64)];
    else if (j == 128) s_pre[128] = states2[b] * (1.0f / 200.0f);
    s_h0[j] = h0[b * 256 + j];
    __syncthreads();

    // layer 1a: relu(pre @ W1a + b1a), W1a is (129,256) row-major
    float acc = b1a[j];
    for (int i = 0; i < 129; ++i) acc = fmaf(s_pre[i], W1a[i * 256 + j], acc);
    s_x[j] = fmaxf(acc, 0.0f);
    __syncthreads();

    // layer 1b: x @ W1b + b1b, (256,256)
    float e = b1b[j];
    for (int i = 0; i < 256; ++i) e = fmaf(s_x[i], W1b[i * 256 + j], e);
    s_emb[j] = e;
    __syncthreads();

    // LSTM cell: gates = emb @ W_ih + h0 @ W_hh + b; gate order i,f,g,o
    float gi = b_ih[j]       + b_hh[j];
    float gf = b_ih[256 + j] + b_hh[256 + j];
    float gg = b_ih[512 + j] + b_hh[512 + j];
    float go = b_ih[768 + j] + b_hh[768 + j];
    for (int i = 0; i < 256; ++i) {
        const float xe = s_emb[i];
        const float hh = s_h0[i];
        const float* wi = W_ih + i * 1024 + j;
        const float* wh = W_hh + i * 1024 + j;
        gi = fmaf(xe, wi[0],   gi); gi = fmaf(hh, wh[0],   gi);
        gf = fmaf(xe, wi[256], gf); gf = fmaf(hh, wh[256], gf);
        gg = fmaf(xe, wi[512], gg); gg = fmaf(hh, wh[512], gg);
        go = fmaf(xe, wi[768], go); go = fmaf(hh, wh[768], go);
    }
    const float cn = sigm(gf) * c0[b * 256 + j] + sigm(gi) * tanhf(gg);
    const float hn = sigm(go) * tanhf(cn);
    out_h1[b * 256 + j] = hn;
    out_c1[b * 256 + j] = cn;
    s_h1[j] = hn;
    __syncthreads();

    // MLP3 layer a: relu(h1 @ W3a + b3a), (256,128)
    if (j < 128) {
        float t = b3a[j];
        for (int i = 0; i < 256; ++i) t = fmaf(s_h1[i], W3a[i * 128 + j], t);
        s_t[j] = fmaxf(t, 0.0f);
    }
    __syncthreads();

    // MLP3 layer b: t @ W3b + b3b, (128,64) -> action embedding
    if (j < 64) {
        float v = b3b[j];
        for (int i = 0; i < 128; ++i) v = fmaf(s_t[i], W3b[i * 64 + j], v);
        actemb[b * 64 + j] = v;
    }
}

// ---------------------------------------------------------------------------
// Kernel 2: candidate scoring with f16 WMMA.
// One block (128 threads = 4 waves) per batch row b; each wave scores
// 8 groups of 16 candidates (4*8*16 = 512 = K).
//
// Layer1: X[16,64]  @ W2a[64,128]  via 2 K-chunks x 8 N-tiles of
//         v_wmma_f32_16x16x32_f16
// Layer2: Y1[16,128] @ W2b[128,64] via 4 K-chunks x 4 N-tiles
// then per-candidate dot with actemb / 8.
//
// Weights are pre-swizzled into exact B-fragment lane/element order in LDS
// (one contiguous 32B read per v16h operand).  Y1 is written to a per-wave
// LDS staging area already in layer-2 A-fragment order.
// ---------------------------------------------------------------------------
__global__ void __launch_bounds__(128) cand_score_kernel(
    const int* __restrict__ cand, const float* __restrict__ rel_emb,
    const float* __restrict__ W2a, const float* __restrict__ b2a,
    const float* __restrict__ W2b, const float* __restrict__ b2b,
    const float* __restrict__ actemb, float* __restrict__ values)
{
    // [kchunk][ntile][lane][elem]
    __shared__ __attribute__((aligned(32))) _Float16 sW1[2][8][32][16]; // 16 KB
    __shared__ __attribute__((aligned(32))) _Float16 sW2[4][4][32][16]; // 16 KB
    __shared__ __attribute__((aligned(32))) _Float16 sY1[4][4][32][16]; // 16 KB (per-wave)
    __shared__ float sB1[128];
    __shared__ float sB2[64];

    const int tid = threadIdx.x;
    const int b   = blockIdx.x;

    // --- swizzle W2a (64x128) into layer-1 B fragments ---
    for (int idx = tid; idx < 8192; idx += 128) {
        const int e    = idx & 15;
        const int ln   = (idx >> 4) & 31;
        const int nt   = (idx >> 9) & 7;
        const int kc   = idx >> 12;
        const int h    = ln >> 4;
        const int n    = nt * 16 + (ln & 15);
        const int k    = kc * 32 + h * 8 + (e & 7) + ((e >> 3) << 4);
        sW1[kc][nt][ln][e] = (_Float16)W2a[k * 128 + n];
    }
    // --- swizzle W2b (128x64) into layer-2 B fragments ---
    for (int idx = tid; idx < 8192; idx += 128) {
        const int e    = idx & 15;
        const int ln   = (idx >> 4) & 31;
        const int nt   = (idx >> 9) & 3;
        const int kc   = (idx >> 11) & 3;
        const int h    = ln >> 4;
        const int n    = nt * 16 + (ln & 15);
        const int k    = kc * 32 + h * 8 + (e & 7) + ((e >> 3) << 4);
        sW2[kc][nt][ln][e] = (_Float16)W2b[k * 64 + n];
    }
    if (tid < 128) sB1[tid] = b2a[tid];
    if (tid < 64)  sB2[tid] = b2b[tid];
    __syncthreads();

    const int wave = tid >> 5;
    const int lane = tid & 31;
    const int hi   = lane >> 4;  // lane half (selects K sub-range / M+8 rows)
    const int lm   = lane & 15;  // N / M position within tile

    float act[4];
#pragma unroll
    for (int nt = 0; nt < 4; ++nt) act[nt] = actemb[b * 64 + nt * 16 + lm];

    for (int g = 0; g < 8; ++g) {
        const int group = wave * 8 + g;
        const int cidx  = cand[b * K_SZ + group * 16 + lm];
        const float* row = rel_emb + (long long)cidx * 64;

        // Layer-1 A fragments (16x32 f16 per chunk) straight from the gather:
        // lane holds K = {kc*32+hi*8 .. +7} and {kc*32+16+hi*8 .. +7}
        v16h a1f[2];
#pragma unroll
        for (int kc = 0; kc < 2; ++kc) {
            const float* p0 = row + kc * 32 + hi * 8;
            const float* p1 = row + kc * 32 + 16 + hi * 8;
            const float4 f0 = *(const float4*)(p0);
            const float4 f1 = *(const float4*)(p0 + 4);
            const float4 f2 = *(const float4*)(p1);
            const float4 f3 = *(const float4*)(p1 + 4);
            a1f[kc][0]  = (_Float16)f0.x; a1f[kc][1]  = (_Float16)f0.y;
            a1f[kc][2]  = (_Float16)f0.z; a1f[kc][3]  = (_Float16)f0.w;
            a1f[kc][4]  = (_Float16)f1.x; a1f[kc][5]  = (_Float16)f1.y;
            a1f[kc][6]  = (_Float16)f1.z; a1f[kc][7]  = (_Float16)f1.w;
            a1f[kc][8]  = (_Float16)f2.x; a1f[kc][9]  = (_Float16)f2.y;
            a1f[kc][10] = (_Float16)f2.z; a1f[kc][11] = (_Float16)f2.w;
            a1f[kc][12] = (_Float16)f3.x; a1f[kc][13] = (_Float16)f3.y;
            a1f[kc][14] = (_Float16)f3.z; a1f[kc][15] = (_Float16)f3.w;
        }

        // Layer 1: 8 N-tiles, bias+ReLU, store swizzled into layer-2 A order
#pragma unroll
        for (int nt = 0; nt < 8; ++nt) {
            v8f acc = {};
            const v16h bf0 = *(const v16h*)&sW1[0][nt][lane][0];
            const v16h bf1 = *(const v16h*)&sW1[1][nt][lane][0];
            acc = __builtin_amdgcn_wmma_f32_16x16x32_f16(
                false, a1f[0], false, bf0, (short)0, acc, false, false);
            acc = __builtin_amdgcn_wmma_f32_16x16x32_f16(
                false, a1f[1], false, bf1, (short)0, acc, false, false);
            const float bias = sB1[nt * 16 + lm];
            const int k2     = nt >> 1;
            const int localK = ((nt & 1) << 4) + lm;     // K index within chunk
            const int lg     = (localK >> 3) & 1;        // target lane half
            const int e      = (localK & 7) + ((localK >> 4) << 3);
#pragma unroll
            for (int r = 0; r < 8; ++r) {
                const int m  = r + (hi << 3);            // candidate row in tile
                const float y = fmaxf(acc[r] + bias, 0.0f);
                sY1[wave][k2][m + (lg << 4)][e] = (_Float16)y;
            }
        }

        // Layer-2 A fragments: contiguous per-lane reads of our own staging
        v16h a2f[4];
#pragma unroll
        for (int k2 = 0; k2 < 4; ++k2)
            a2f[k2] = *(const v16h*)&sY1[wave][k2][lane][0];

        // Layer 2 + fused dot with action embedding
        float part[8];
#pragma unroll
        for (int r = 0; r < 8; ++r) part[r] = 0.0f;
#pragma unroll
        for (int nt = 0; nt < 4; ++nt) {
            v8f acc = {};
#pragma unroll
            for (int k2 = 0; k2 < 4; ++k2) {
                const v16h bf = *(const v16h*)&sW2[k2][nt][lane][0];
                acc = __builtin_amdgcn_wmma_f32_16x16x32_f16(
                    false, a2f[k2], false, bf, (short)0, acc, false, false);
            }
            const float bias = sB2[nt * 16 + lm];
            const float w    = act[nt];
#pragma unroll
            for (int r = 0; r < 8; ++r) part[r] += (acc[r] + bias) * w;
        }

        // reduce over the 16 lanes sharing a candidate (butterfly, wave32)
#pragma unroll
        for (int r = 0; r < 8; ++r) {
            part[r] += __shfl_xor(part[r], 1, 32);
            part[r] += __shfl_xor(part[r], 2, 32);
            part[r] += __shfl_xor(part[r], 4, 32);
            part[r] += __shfl_xor(part[r], 8, 32);
        }
        if (lm == 0) {
#pragma unroll
            for (int r = 0; r < 8; ++r)
                values[b * K_SZ + group * 16 + r + (hi << 3)] = part[r] * 0.125f;
        }
    }
}

// ---------------------------------------------------------------------------
extern "C" void kernel_launch(void* const* d_in, const int* in_sizes, int n_in,
                              void* d_out, int out_size, void* d_ws, size_t ws_size,
                              hipStream_t stream) {
    const float* states2  = (const float*)d_in[0];
    const float* h0       = (const float*)d_in[1];
    const float* c0       = (const float*)d_in[2];
    const int*   actions0 = (const int*)  d_in[3];
    const int*   actions1 = (const int*)  d_in[4];
    const int*   cand     = (const int*)  d_in[5];
    const float* node_emb = (const float*)d_in[6];
    const float* rel_emb  = (const float*)d_in[7];
    const float* W1a = (const float*)d_in[8];
    const float* b1a = (const float*)d_in[9];
    const float* W1b = (const float*)d_in[10];
    const float* b1b = (const float*)d_in[11];
    const float* W2a = (const float*)d_in[12];
    const float* b2a = (const float*)d_in[13];
    const float* W2b = (const float*)d_in[14];
    const float* b2b = (const float*)d_in[15];
    const float* W3a = (const float*)d_in[16];
    const float* b3a = (const float*)d_in[17];
    const float* W3b = (const float*)d_in[18];
    const float* b3b = (const float*)d_in[19];
    const float* W_ih = (const float*)d_in[20];
    const float* W_hh = (const float*)d_in[21];
    const float* b_ih = (const float*)d_in[22];
    const float* b_hh = (const float*)d_in[23];

    float* out    = (float*)d_out;
    float* values = out;                                 // [1024,512]
    float* out_h1 = out + B_SZ * K_SZ;                   // [1024,256]
    float* out_c1 = out + B_SZ * K_SZ + B_SZ * 256;      // [1024,256]
    float* actemb = (float*)d_ws;                        // [1024,64] scratch

    hipLaunchKernelGGL(action_path_kernel, dim3(B_SZ), dim3(256), 0, stream,
        states2, h0, c0, actions0, actions1, node_emb,
        W1a, b1a, W1b, b1b, W3a, b3a, W3b, b3b,
        W_ih, W_hh, b_ih, b_hh,
        out_h1, out_c1, actemb);

    hipLaunchKernelGGL(cand_score_kernel, dim3(B_SZ), dim3(128), 0, stream,
        cand, rel_emb, W2a, b2a, W2b, b2b, actemb, values);
}
